// MambaSSM_52905407152479
// MI455X (gfx1250) — matrix operand order repaired and checked by reading, not compiled, not run on hardware
//
#include <hip/hip_runtime.h>
#include <hip/hip_bf16.h>
#include <math.h>

// ---------------------------------------------------------------------------
// Mamba SSM forward for MI455X (gfx1250, wave32, WMMA)
// Sizes fixed by the reference: B=4, L=2048, d_model=1024, d_state=16,
// d_conv=4, expand=2 -> d_inner=2048, M = B*L = 8192.
// ---------------------------------------------------------------------------

typedef __attribute__((ext_vector_type(2)))  __bf16 bf16x2;
typedef __attribute__((ext_vector_type(4)))  __bf16 bf16x4;
typedef __attribute__((ext_vector_type(8)))  __bf16 bf16x8;
typedef __attribute__((ext_vector_type(16))) __bf16 bf16x16;
typedef __attribute__((ext_vector_type(8)))  float  floatx8;

#define DM   1024          // d_model (K of GEMM1)
#define DI   2048          // d_inner
#define NS   16            // d_state
#define BATCH 4
#define LSEQ 2048
#define MTOT (BATCH*LSEQ)  // 8192
#define LDSS 48            // LDS row stride in halfs (96B, 16B-aligned rows)

__device__ __forceinline__ __bf16 f2bf(float f) {
    unsigned int u = __builtin_bit_cast(unsigned int, f);
    unsigned int r = (u + 0x7FFFu + ((u >> 16) & 1u)) >> 16;
    return __builtin_bit_cast(__bf16, (unsigned short)r);
}

// Packed fp32x2 -> bf16x2 (RNE).  Native v_cvt_pk_bf16_f32 when available.
#if __has_builtin(__builtin_amdgcn_cvt_pk_bf16_f32)
__device__ __forceinline__ bf16x2 f2bf2(float a, float b) {
    return __builtin_amdgcn_cvt_pk_bf16_f32(a, b);
}
#else
__device__ __forceinline__ bf16x2 f2bf2(float a, float b) {
    bf16x2 r; r.x = f2bf(a); r.y = f2bf(b); return r;
}
#endif

__device__ __forceinline__ bf16x4 f2bf4(float4 v) {
    bf16x2 lo = f2bf2(v.x, v.y);
    bf16x2 hi = f2bf2(v.z, v.w);
    return __builtin_shufflevector(lo, hi, 0, 1, 2, 3);
}

__device__ __forceinline__ bf16x16 cat16(bf16x8 lo, bf16x8 hi) {
    return __builtin_shufflevector(lo, hi, 0,1,2,3,4,5,6,7,8,9,10,11,12,13,14,15);
}

// Load A fragment (16x32 bf16): lane l, row = l%16; K = 8*(l/16)+0..7 and 16+8*(l/16)+0..7
__device__ __forceinline__ bf16x16 load_a_frag(const __bf16* lds, int rowbase, int l16, int half) {
    const __bf16* p = lds + (rowbase + l16) * LDSS + half * 8;
    bf16x8 lo = *(const bf16x8*)p;
    bf16x8 hi = *(const bf16x8*)(p + 16);
    return cat16(lo, hi);
}

// Load B fragment (32x16 bf16) from n-major LDS [n][k]: lane l, col = l%16; K = 8*(l/16)+0..7 and 16+8*(l/16)+0..7
__device__ __forceinline__ bf16x16 load_b_frag(const __bf16* lds, int colbase, int l16, int half) {
    const __bf16* p = lds + (colbase + l16) * LDSS + half * 8;
    bf16x8 lo = *(const bf16x8*)p;
    bf16x8 hi = *(const bf16x8*)(p + 16);
    return cat16(lo, hi);
}

// ---------------------------------------------------------------------------
// Kernel 1: x_in = silu(x @ Wg) * (x @ Wi), gate/in fused in one pass.
// grid = (DI/64, MTOT/128), block = 256 (8 waves; wave (wm=w>>1, wn=w&1))
// Each wave: 32x32 gate tile + 32x32 in tile (2x2 WMMA tiles each).
// ---------------------------------------------------------------------------
__global__ __launch_bounds__(256) void k1_gemm_gate(const float* __restrict__ x,
                                                    const float* __restrict__ W_in,
                                                    float* __restrict__ x_in) {
    __shared__ __align__(16) __bf16 Alds[128 * LDSS];
    __shared__ __align__(16) __bf16 Bglds[64 * LDSS];
    __shared__ __align__(16) __bf16 Bilds[64 * LDSS];

    const int tid  = threadIdx.x;
    const int l16  = tid & 15;
    const int half = (tid >> 4) & 1;
    const int wave = tid >> 5;
    const int wm   = wave >> 1;        // 0..3
    const int wn   = wave & 1;         // 0..1
    const int m0   = blockIdx.y * 128;
    const int ng0  = blockIdx.x * 64;  // gate col base; in cols at ng0+DI

    const floatx8 vz = {0.f,0.f,0.f,0.f,0.f,0.f,0.f,0.f};
    floatx8 accg[2][2], acci[2][2];
#pragma unroll
    for (int a = 0; a < 2; ++a)
#pragma unroll
        for (int b = 0; b < 2; ++b) { accg[a][b] = vz; acci[a][b] = vz; }

    const int c4 = tid & 7;      // A: k-group (4 floats)
    const int r0 = tid >> 3;     // A: base row

    for (int k0 = 0; k0 < DM; k0 += 32) {
        // ---- stage A tile 128x32 (fp32 -> bf16, packed 8B LDS stores) ----
#pragma unroll
        for (int q = 0; q < 4; ++q) {
            const int row = r0 + q * 32;
            const float4 v = *(const float4*)(x + (size_t)(m0 + row) * DM + k0 + c4 * 4);
            *(bf16x4*)(Alds + row * LDSS + c4 * 4) = f2bf4(v);
        }
        // ---- stage B tiles (gate cols + in cols), transposed to n-major ----
#pragma unroll
        for (int j = 0; j < 4; ++j) {
            const int idx = tid + j * 256;       // 0..1023
            const int s   = idx >> 9;            // 0=gate, 1=in
            const int r   = idx & 511;
            const int k   = r >> 4;              // 0..31
            const int n4  = r & 15;              // 0..15 (groups of 4 cols)
            const int nb  = ng0 + (s ? DI : 0);
            const float4 v = *(const float4*)(W_in + (size_t)(k0 + k) * (2 * DI) + nb + n4 * 4);
            const bf16x4 b4 = f2bf4(v);
            __bf16* dst = (s ? Bilds : Bglds) + (n4 * 4) * LDSS + k;
            dst[0 * LDSS] = b4.x; dst[1 * LDSS] = b4.y;
            dst[2 * LDSS] = b4.z; dst[3 * LDSS] = b4.w;
        }
        // prefetch next K chunk while this one computes
        if (k0 + 32 < DM) {
            __builtin_prefetch(x + (size_t)(m0 + r0) * DM + (k0 + 32) + c4 * 4, 0, 0);
            __builtin_prefetch(W_in + (size_t)(k0 + 32 + (tid >> 4)) * (2 * DI) + ng0 + (tid & 15) * 4, 0, 0);
        }
        __syncthreads();

        bf16x16 afr[2], bgf[2], bif[2];
#pragma unroll
        for (int rt = 0; rt < 2; ++rt)
            afr[rt] = load_a_frag(Alds, wm * 32 + rt * 16, l16, half);
#pragma unroll
        for (int ct = 0; ct < 2; ++ct) {
            bgf[ct] = load_b_frag(Bglds, wn * 32 + ct * 16, l16, half);
            bif[ct] = load_b_frag(Bilds, wn * 32 + ct * 16, l16, half);
        }
#pragma unroll
        for (int rt = 0; rt < 2; ++rt)
#pragma unroll
            for (int ct = 0; ct < 2; ++ct) {
                accg[rt][ct] = __builtin_amdgcn_wmma_f32_16x16x32_bf16(
                    false, afr[rt], false, bgf[ct], (short)0, accg[rt][ct], false, false);
                acci[rt][ct] = __builtin_amdgcn_wmma_f32_16x16x32_bf16(
                    false, afr[rt], false, bif[ct], (short)0, acci[rt][ct], false, false);
            }
        __syncthreads();
    }

    // epilogue: x_in = silu(gate) * in
#pragma unroll
    for (int rt = 0; rt < 2; ++rt)
#pragma unroll
        for (int ct = 0; ct < 2; ++ct)
#pragma unroll
            for (int v = 0; v < 8; ++v) {
                const int m = m0 + wm * 32 + rt * 16 + v + half * 8;
                const int n = ng0 + wn * 32 + ct * 16 + l16;
                const float g  = accg[rt][ct][v];
                const float xi = acci[rt][ct][v];
                const float sg = g / (1.0f + __expf(-g));
                x_in[(size_t)m * DI + n] = sg * xi;
            }
}

// ---------------------------------------------------------------------------
// Kernel 2: causal depthwise conv width-4 + bias.  xc[b,t,c].
// ---------------------------------------------------------------------------
__global__ __launch_bounds__(256) void k2_conv(const float* __restrict__ x_in,
                                               const float* __restrict__ conv_w,
                                               const float* __restrict__ conv_b,
                                               float* __restrict__ xc) {
    const int gid = blockIdx.x * 256 + threadIdx.x;   // over MTOT * DI/4
    const int c4  = gid & 511;                        // float4 group in channel dim
    const int m   = gid >> 9;
    const int c   = c4 * 4;
    const int t   = m & (LSEQ - 1);

    float w0[4], w1[4], w2[4], w3[4];
    *(float4*)w0 = *(const float4*)(conv_w + (size_t)(c + 0) * 4);
    *(float4*)w1 = *(const float4*)(conv_w + (size_t)(c + 1) * 4);
    *(float4*)w2 = *(const float4*)(conv_w + (size_t)(c + 2) * 4);
    *(float4*)w3 = *(const float4*)(conv_w + (size_t)(c + 3) * 4);

    float4 r;
    r.x = conv_b[c]; r.y = conv_b[c + 1]; r.z = conv_b[c + 2]; r.w = conv_b[c + 3];
#pragma unroll
    for (int j = 0; j < 4; ++j) {
        const int tj = t - 3 + j;
        if (tj >= 0) {
            const float4 xv = *(const float4*)(x_in + (size_t)(m - 3 + j) * DI + c);
            r.x += w0[j] * xv.x; r.y += w1[j] * xv.y;
            r.z += w2[j] * xv.z; r.w += w3[j] * xv.w;
        }
    }
    *(float4*)(xc + (size_t)m * DI + c) = r;
}

// ---------------------------------------------------------------------------
// Kernel 3: delta/B/C projections + xc@D via one WMMA pass (64-col packed B:
// [W_delta | W_B | W_C | D | zeros]).  Epilogue fuses A_d=exp(delta*logA),
// B_d=delta*Bm.  grid = MTOT/128, block = 256 (8 waves, 16 rows each).
// ---------------------------------------------------------------------------
__global__ __launch_bounds__(256) void k3_proj(const float* __restrict__ xc,
                                               const float* __restrict__ W_delta,
                                               const float* __restrict__ b_delta,
                                               const float* __restrict__ W_B,
                                               const float* __restrict__ W_C,
                                               const float* __restrict__ Dvec,
                                               float* __restrict__ Ad,
                                               float* __restrict__ Bd,
                                               float* __restrict__ Cmo,
                                               float* __restrict__ dsum) {
    __shared__ __align__(16) __bf16 Alds[128 * LDSS];
    __shared__ __align__(16) __bf16 Wlds[64 * LDSS];

    const int tid  = threadIdx.x;
    const int l16  = tid & 15;
    const int half = (tid >> 4) & 1;
    const int wave = tid >> 5;
    const int m0   = blockIdx.x * 128;

    const floatx8 vz = {0.f,0.f,0.f,0.f,0.f,0.f,0.f,0.f};
    floatx8 acc[4] = {vz, vz, vz, vz};

    const int c4 = tid & 7;
    const int r0 = tid >> 3;
    const int wn = tid & 63;     // packed col for W staging
    const int kq = tid >> 6;     // 0..3

    for (int k0 = 0; k0 < DI; k0 += 32) {
        // ---- stage A (xc rows) ----
#pragma unroll
        for (int q = 0; q < 4; ++q) {
            const int row = r0 + q * 32;
            const float4 v = *(const float4*)(xc + (size_t)(m0 + row) * DI + k0 + c4 * 4);
            *(bf16x4*)(Alds + row * LDSS + c4 * 4) = f2bf4(v);
        }
        // ---- stage packed W (n-major) ----
#pragma unroll
        for (int j = 0; j < 8; ++j) {
            const int k  = kq * 8 + j;
            const int kk = k0 + k;
            float v;
            if      (wn < 16) v = W_delta[(size_t)kk * NS + wn];
            else if (wn < 32) v = W_B[(size_t)kk * NS + (wn - 16)];
            else if (wn < 48) v = W_C[(size_t)kk * NS + (wn - 32)];
            else if (wn == 48) v = Dvec[kk];
            else              v = 0.0f;
            Wlds[wn * LDSS + k] = f2bf(v);
        }
        if (k0 + 32 < DI)
            __builtin_prefetch(xc + (size_t)(m0 + r0) * DI + (k0 + 32) + c4 * 4, 0, 0);
        __syncthreads();

        const bf16x16 afr = load_a_frag(Alds, wave * 16, l16, half);
#pragma unroll
        for (int ct = 0; ct < 4; ++ct) {
            const bf16x16 bfr = load_b_frag(Wlds, ct * 16, l16, half);
            acc[ct] = __builtin_amdgcn_wmma_f32_16x16x32_bf16(
                false, afr, false, bfr, (short)0, acc[ct], false, false);
        }
        __syncthreads();
    }

    // epilogue: discretize
    const int   n  = l16;
    const float lA = -0.5f * log1pf((float)(n + 1) * (1.0f / (float)NS));
    const float bb = b_delta[n];
#pragma unroll
    for (int v = 0; v < 8; ++v) {
        const int m = m0 + wave * 16 + v + half * 8;
        const float delta = acc[0][v] + bb;
        const float a_d   = __expf(delta * lA);
        const float bm    = acc[1][v];
        const float cm    = acc[2][v];
        const size_t o    = (size_t)m * NS + n;
        Ad[o]  = a_d;
        Bd[o]  = delta * bm;
        Cmo[o] = cm;
        if (n == 0) dsum[m] = acc[3][v];   // column 48 of packed B == xc @ D
    }
}

// ---------------------------------------------------------------------------
// Kernel 4: chunked parallel scan h_t = a_t h_{t-1} + b_t, then
// y = sum_n(Cm*h) + dsum via width-16 shfl_xor butterflies.
// grid = BATCH, block = 256 (16 states x 16 chunks of 128 steps).
// ---------------------------------------------------------------------------
__global__ __launch_bounds__(256) void k4_scan(const float* __restrict__ Ad,
                                               const float* __restrict__ Bd,
                                               const float* __restrict__ Cmi,
                                               const float* __restrict__ dsum,
                                               float* __restrict__ y) {
    __shared__ float ap_s[16][16], he_s[16][16], hi_s[16][16];
    const int b   = blockIdx.x;
    const int tid = threadIdx.x;
    const int n   = tid & 15;
    const int c   = tid >> 4;
    const int CH  = LSEQ / 16;                 // 128
    const size_t base = (size_t)b * LSEQ * NS;

    // pass 1: chunk-local scan -> (prod a, h_end)
    float h = 0.0f, ap = 1.0f;
    for (int i = 0; i < CH; ++i) {
        const size_t o = base + (size_t)(c * CH + i) * NS + n;
        const float a = Ad[o];
        h  = a * h + Bd[o];
        ap *= a;
    }
    ap_s[c][n] = ap; he_s[c][n] = h;
    __syncthreads();

    // pass 2: exclusive scan over chunks (one lane per state)
    if (tid < 16) {
        float carry = 0.0f;
        for (int cc = 0; cc < 16; ++cc) {
            hi_s[cc][tid] = carry;
            carry = ap_s[cc][tid] * carry + he_s[cc][tid];
        }
    }
    __syncthreads();

    // pass 3: re-scan with carry-in, reduce over states, emit y
    h = hi_s[c][n];
    for (int i = 0; i < CH; ++i) {
        const int t = c * CH + i;
        const size_t o = base + (size_t)t * NS + n;
        h = Ad[o] * h + Bd[o];
        float part = Cmi[o] * h;
        part += __shfl_xor(part, 1, 16);
        part += __shfl_xor(part, 2, 16);
        part += __shfl_xor(part, 4, 16);
        part += __shfl_xor(part, 8, 16);
        if (n == 0) y[b * LSEQ + t] = part + dsum[b * LSEQ + t];
    }
}

// ---------------------------------------------------------------------------
// Kernel 5: out[b,t,:] = y[b,t] * w_out
// ---------------------------------------------------------------------------
__global__ __launch_bounds__(256) void k5_outer(const float* __restrict__ y,
                                                const float* __restrict__ w_out,
                                                float* __restrict__ out) {
    const int gid = blockIdx.x * 256 + threadIdx.x;    // over MTOT * DM/4
    const int d4  = gid & 255;
    const int m   = gid >> 8;
    const float  yv = y[m];
    const float4 w  = ((const float4*)w_out)[d4];
    float4 r; r.x = yv * w.x; r.y = yv * w.y; r.z = yv * w.z; r.w = yv * w.w;
    ((float4*)out)[(size_t)m * 256 + d4] = r;
}

// ---------------------------------------------------------------------------
extern "C" void kernel_launch(void* const* d_in, const int* in_sizes, int n_in,
                              void* d_out, int out_size, void* d_ws, size_t ws_size,
                              hipStream_t stream) {
    const float* x       = (const float*)d_in[0];
    const float* W_in    = (const float*)d_in[1];
    const float* conv_w  = (const float*)d_in[2];
    const float* conv_b  = (const float*)d_in[3];
    const float* W_delta = (const float*)d_in[4];
    const float* b_delta = (const float*)d_in[5];
    const float* W_B     = (const float*)d_in[6];
    const float* W_C     = (const float*)d_in[7];
    const float* Dvec    = (const float*)d_in[8];
    const float* w_out   = (const float*)d_in[9];
    float* out = (float*)d_out;

    float* ws   = (float*)d_ws;
    float* x_in = ws;                              // MTOT*DI
    float* xc   = x_in + (size_t)MTOT * DI;        // MTOT*DI
    float* Ad   = xc   + (size_t)MTOT * DI;        // MTOT*NS
    float* Bd   = Ad   + (size_t)MTOT * NS;
    float* Cm   = Bd   + (size_t)MTOT * NS;
    float* dsum = Cm   + (size_t)MTOT * NS;        // MTOT
    float* yv   = dsum + MTOT;                     // MTOT

    // 1) fused gate GEMM: x_in = silu(x@Wg) * (x@Wi)
    k1_gemm_gate<<<dim3(DI / 64, MTOT / 128), 256, 0, stream>>>(x, W_in, x_in);
    // 2) causal depthwise conv
    k2_conv<<<(MTOT * (DI / 4)) / 256, 256, 0, stream>>>(x_in, conv_w, conv_b, xc);
    // 3) projections + discretization (WMMA, packed 64-col B)
    k3_proj<<<MTOT / 128, 256, 0, stream>>>(xc, W_delta, b_delta, W_B, W_C, Dvec,
                                            Ad, Bd, Cm, dsum);
    // 4) parallel scan + state reduction
    k4_scan<<<BATCH, 256, 0, stream>>>(Ad, Bd, Cm, dsum, yv);
    // 5) outer product with w_out
    k5_outer<<<(MTOT * (DM / 4)) / 256, 256, 0, stream>>>(yv, w_out, out);
}
